// Relation_Net_86715389706350
// MI455X (gfx1250) — matrix operand order repaired and checked by reading, not compiled, not run on hardware
//
#include <hip/hip_runtime.h>

typedef __attribute__((ext_vector_type(16))) __bf16 v16bf;
typedef __attribute__((ext_vector_type(8)))  float  v8f;

#define DDIM 200
#define OUTK 5
#define KCH 7              // ceil(200/32) K-chunks (pad to 224)
#define JT 13              // ceil(200/16) N-tiles  (pad to 208)
#define MTILE 128
#define R1_STRIDE_B 448    // 224 bf16 * 2 bytes per row
#define R2_STRIDE_F 208    // floats per row

// ---------------------------------------------------------------------------
// Prep: repack W (f32, [5][200][200]) into bf16 WMMA B-fragments.
// Fragment f = (k*JT + jt)*KCH + kc is a 32x16 (KxN) bf16 tile, 1024 bytes,
// laid out lane-major: lane l holds 16 contiguous halves at byte l*32,
// half h <-> K = 16*(l/16) + h, column n = l%16  (contiguous K order, same as the A tile).
// Out-of-range (i>=200 or j>=200) elements are zero (padding).
// ---------------------------------------------------------------------------
__global__ void prep_w_kernel(const float* __restrict__ W, __bf16* __restrict__ wb) {
    int idx = blockIdx.x * blockDim.x + threadIdx.x;
    const int total = OUTK * JT * KCH * 512;
    if (idx >= total) return;
    int e  = idx & 511;            // element within fragment (lane*16 + half)
    int f  = idx >> 9;             // fragment index
    int kc = f % KCH;
    int jt = (f / KCH) % JT;
    int k  = f / (KCH * JT);
    int lane = e >> 4;
    int half = e & 15;
    int kk = ((lane >> 4) << 4) + half;   // K within chunk
    int n  = lane & 15;                   // column within tile
    int i = kc * 32 + kk;
    int j = jt * 16 + n;
    float v = 0.0f;
    if (i < DDIM && j < DDIM) v = W[k * (DDIM * DDIM) + i * DDIM + j];
    wb[idx] = (__bf16)v;
}

// ---------------------------------------------------------------------------
// Fused relu -> (r1 x Wk) GEMM (bf16 WMMA, f32 acc) -> elementwise * relu(r2)
// -> reduce over j -> + bias.  One WG = 128 batch rows, one wave = 16 rows.
// ---------------------------------------------------------------------------
__global__ __launch_bounds__(256)
void relnet_kernel(const float* __restrict__ x1, const float* __restrict__ x2,
                   const float* __restrict__ bias, const __bf16* __restrict__ wb,
                   float* __restrict__ out) {
    extern __shared__ char smem[];
    char*  r1s = smem;                                     // bf16, stride 448 B/row
    float* r2s = (float*)(smem + MTILE * R1_STRIDE_B);     // f32,  stride 208 f/row

    const int tid   = threadIdx.x;
    const int bbase = blockIdx.x * MTILE;

    // ---- stage r1: relu -> bf16 into LDS (row-major, K padded to 224) ----
    const float4* x1v = (const float4*)(x1 + (size_t)bbase * DDIM);
    for (int t = tid; t < MTILE * 50; t += 256) {
        int row = t / 50, c4 = t % 50;
        float4 v = x1v[row * 50 + c4];
        union { __bf16 h[4]; uint2 u; } p;
        p.h[0] = (__bf16)fmaxf(v.x, 0.f);
        p.h[1] = (__bf16)fmaxf(v.y, 0.f);
        p.h[2] = (__bf16)fmaxf(v.z, 0.f);
        p.h[3] = (__bf16)fmaxf(v.w, 0.f);
        *(uint2*)(r1s + row * R1_STRIDE_B + c4 * 8) = p.u;
    }
    for (int t = tid; t < MTILE * 6; t += 256) {           // zero-pad K 200..223
        int row = t / 6, q = t % 6;
        *(uint2*)(r1s + row * R1_STRIDE_B + 400 + q * 8) = make_uint2(0u, 0u);
    }

    // ---- stage r2: relu -> f32 into LDS (stride 208, J padded) ----
    const float4* x2v = (const float4*)(x2 + (size_t)bbase * DDIM);
    for (int t = tid; t < MTILE * 50; t += 256) {
        int row = t / 50, c4 = t % 50;
        float4 v = x2v[row * 50 + c4];
        v.x = fmaxf(v.x, 0.f); v.y = fmaxf(v.y, 0.f);
        v.z = fmaxf(v.z, 0.f); v.w = fmaxf(v.w, 0.f);
        ((float4*)(r2s + row * R2_STRIDE_F))[c4] = v;
    }
    for (int t = tid; t < MTILE * 2; t += 256) {           // zero-pad J 200..207
        int row = t / 2, q = t % 2;
        ((float4*)(r2s + row * R2_STRIDE_F + 200))[q] = make_float4(0.f, 0.f, 0.f, 0.f);
    }
    __syncthreads();

    const int lane = tid & 31;
    const int wave = tid >> 5;

    // ---- load this wave's 16x224 A-tile into registers (contiguous K order) ----
    // lane m / lane 16+m both hold row m; lanes<16 take K {0..15} of each
    // 32-chunk, lanes>=16 take K {16..31}.
    const int  mrow = wave * 16 + (lane & 15);
    const char* arow = r1s + mrow * R1_STRIDE_B + ((lane & 16) ? 32 : 0);
    v16bf a[KCH];
#pragma unroll
    for (int kc = 0; kc < KCH; ++kc) {
        union { int4 q[2]; v16bf v; } ua;
        ua.q[0] = *(const int4*)(arow + kc * 64);
        ua.q[1] = *(const int4*)(arow + kc * 64 + 16);
        a[kc] = ua.v;
    }

    // r2 base for epilogue: C layout -> lanes 0-15: rows v, lanes 16-31: rows 8+v,
    // column = jt*16 + lane%16.
    const float* r2lane = r2s + (wave * 16 + ((lane >> 4) << 3)) * R2_STRIDE_F + (lane & 15);

    const int4* wbv = (const int4*)wb;                     // 64 int4 per fragment

    for (int k = 0; k < OUTK; ++k) {
        float acc[8];
#pragma unroll
        for (int v = 0; v < 8; ++v) acc[v] = 0.f;

        for (int jt = 0; jt < JT; ++jt) {
            v8f c = {};
            const int fbase = ((k * JT + jt) * KCH) * 64 + lane * 2;
#pragma unroll
            for (int kc = 0; kc < KCH; ++kc) {
                union { int4 q[2]; v16bf v; } ub;
                ub.q[0] = wbv[fbase + kc * 64];
                ub.q[1] = wbv[fbase + kc * 64 + 1];
                c = __builtin_amdgcn_wmma_f32_16x16x32_bf16(
                        false, a[kc], false, ub.v, (short)0, c, false, false);
            }
            const float* r2p = r2lane + jt * 16;
#pragma unroll
            for (int v = 0; v < 8; ++v)
                acc[v] = fmaf(c[v], r2p[v * R2_STRIDE_F], acc[v]);
        }

        // reduce across the 16 lanes of each half (rows stay separated)
#pragma unroll
        for (int v = 0; v < 8; ++v) {
            float s = acc[v];
            s += __shfl_xor(s, 1);
            s += __shfl_xor(s, 2);
            s += __shfl_xor(s, 4);
            s += __shfl_xor(s, 8);
            acc[v] = s;
        }
        if ((lane & 15) == 0) {
            float bk = bias[k];
            int rbase = bbase + wave * 16 + ((lane >> 4) << 3);
#pragma unroll
            for (int v = 0; v < 8; ++v)
                out[(rbase + v) * OUTK + k] = acc[v] + bk;
        }
    }
}

extern "C" void kernel_launch(void* const* d_in, const int* in_sizes, int n_in,
                              void* d_out, int out_size, void* d_ws, size_t ws_size,
                              hipStream_t stream) {
    const float* x1   = (const float*)d_in[0];   // (65536, 200) f32
    const float* x2   = (const float*)d_in[1];   // (65536, 200) f32
    const float* W    = (const float*)d_in[2];   // (5, 40000)   f32
    const float* bias = (const float*)d_in[3];   // (5,)         f32
    float* out = (float*)d_out;                  // (65536, 5)   f32

    const int B = in_sizes[0] / DDIM;            // 65536

    __bf16* wb = (__bf16*)d_ws;                  // 465,920 bytes of repacked W

    const int total = OUTK * JT * KCH * 512;
    prep_w_kernel<<<(total + 255) / 256, 256, 0, stream>>>(W, wb);

    const size_t lds_bytes = (size_t)MTILE * R1_STRIDE_B + (size_t)MTILE * R2_STRIDE_F * 4;
    relnet_kernel<<<B / MTILE, 256, lds_bytes, stream>>>(x1, x2, bias, wb, out);
}
